// SuperPointNet_14929306321123
// MI455X (gfx1250) — matrix operand order repaired and checked by reading, not compiled, not run on hardware
//
#include <hip/hip_runtime.h>
#include <hip/hip_bf16.h>

typedef _Float16 half_t;
typedef __attribute__((ext_vector_type(16))) _Float16 v16h;
typedef __attribute__((ext_vector_type(8)))  float    v8f;

#define THRESH 0.005f
#define NMSR 4
#define MAXK 2048
#define MAXC 16384

// ---------------------------------------------------------------------------
// Weight repack: f32 OIHW -> f16 WMMA A-fragment layout.
// A (16x32 f16): lane l holds row M=l%16; half h holds
//   K = (h<8 ? h : h+8) + (l<16 ? 0 : 8)     (per CDNA5 ISA 16-bit A table)
// Packed index: ((((ocTile*ntaps+tap)*nKc)+kcb)*32 + lane)*16 + h
// ocTile count is padded to a multiple of 2 (32-oc register tiles).
// ---------------------------------------------------------------------------
__global__ void pack_weights_k(const float* __restrict__ w, half_t* __restrict__ ap,
                               int OC, int IC, int KH, int KW, int total) {
  int idx = blockIdx.x * blockDim.x + threadIdx.x;
  if (idx >= total) return;
  int nKc = IC >> 5, ntaps = KH * KW;
  int h = idx & 15;
  int t = idx >> 4;
  int lane = t & 31; t >>= 5;
  int kcb = t % nKc; t /= nKc;
  int tap = t % ntaps;
  int ocT = t / ntaps;
  int k = (h < 8 ? h : h + 8) + (lane < 16 ? 0 : 8);
  int m = lane & 15;
  int oc = (ocT << 4) + m;
  int ic = (kcb << 5) + k;
  int ky = tap / KW, kx = tap % KW;
  float v = (oc < OC) ? w[(((size_t)oc * IC + ic) * KH + ky) * KW + kx] : 0.f;
  ap[idx] = (half_t)v;
}

// ---------------------------------------------------------------------------
// Implicit-GEMM conv (3x3 pad=1 or 1x1) with V_WMMA_F32_16X16X32_F16.
// Fully static shapes. Each wave computes a 2x2 register tile:
//   2 oc-tiles (32 output channels) x 2 row-adjacent pixel tiles (16 px each).
// Per K-step: 2 A loads + 2 B loads feed 4 independent WMMAs (4 acc chains).
// B fragment: lane%16 = pixel N, 16 halves = 16 channels K (A's k order) -> 16
// strided 2-byte loads per lane, zero-filled for x or y outside the image.
// ---------------------------------------------------------------------------
template <int OC, int IC, int H, int W, int K3, int RELU>
__global__ void __launch_bounds__(256)
conv_wmma_k(const half_t* __restrict__ in, const half_t* __restrict__ ap,
            const float* __restrict__ bias, half_t* __restrict__ out16,
            float* __restrict__ out32) {
  constexpr int nKc = IC >> 5;
  constexpr int ntaps = K3 ? 9 : 1;
  constexpr int numXT = W >> 4;
  constexpr int numYT = H >> 1;              // H always even here
  constexpr int numOcT2 = (OC + 31) >> 5;
  constexpr int tilesPerOc = numYT * numXT;
  constexpr int totalTiles = numOcT2 * tilesPerOc;
  constexpr size_t HW = (size_t)H * W;
  constexpr int aStride = ntaps * nKc * 32 * 16;   // halves per 16-oc tile

  int lane = threadIdx.x & 31;
  int wid = blockIdx.x * (blockDim.x >> 5) + (threadIdx.x >> 5);
  if (wid >= totalTiles) return;             // wave-uniform
  int ocT2 = wid / tilesPerOc;
  int rem = wid - ocT2 * tilesPerOc;
  int ty = rem / numXT;
  int x0 = (rem - ty * numXT) << 4;
  int y0 = ty << 1;

  v8f acc00 = {}, acc01 = {}, acc10 = {}, acc11 = {};
  const half_t* apBase0 = ap + (size_t)(2 * ocT2) * aStride;
  const half_t* apBase1 = apBase0 + aStride;

  #pragma unroll
  for (int tap = 0; tap < ntaps; ++tap) {
    const int dy = K3 ? (tap / 3 - 1) : 0;
    const int dx = K3 ? (tap % 3 - 1) : 0;
    int yy0 = y0 + dy;
    int yy1 = yy0 + 1;
    bool v0 = !K3 || (yy0 >= 0 && yy0 < H);  // wave-uniform
    bool v1 = !K3 || (yy1 >= 0 && yy1 < H);
    if (!v0 && !v1) continue;
    int xb = x0 + dx;
    bool interior = (xb >= 0) && (xb + 16 <= W);
    #pragma unroll 2
    for (int kcb = 0; kcb < nKc; ++kcb) {
      size_t aoff = (size_t)((tap * nKc + kcb) * 32 + lane) << 4;
      v16h a0 = *(const v16h*)(apBase0 + aoff);
      v16h a1 = *(const v16h*)(apBase1 + aoff);
      const half_t* row = in + (size_t)(kcb << 5) * HW;
      v16h b0 = {}, b1 = {};
      {
        int x = xb + (lane & 15);
        bool inx = (x >= 0 && x < W);
        int kh = (lane >> 4) << 3;
        #pragma unroll
        for (int s = 0; s < 16; ++s) {
          const half_t* cp = row + (size_t)((s < 8 ? s : s + 8) + kh) * HW;
          if (v0) b0[s] = inx ? cp[(size_t)yy0 * W + x] : (half_t)0.f;
          if (v1) b1[s] = inx ? cp[(size_t)yy1 * W + x] : (half_t)0.f;
        }
      }
      if (kcb + 1 < nKc)                     // gfx1250 global_prefetch path
        __builtin_prefetch(row + 32 * HW + (size_t)(v0 ? yy0 : yy1) * W + x0, 0, 3);
      acc00 = __builtin_amdgcn_wmma_f32_16x16x32_f16(false, a0, false, b0, (short)0, acc00, false, false);
      acc01 = __builtin_amdgcn_wmma_f32_16x16x32_f16(false, a0, false, b1, (short)0, acc01, false, false);
      acc10 = __builtin_amdgcn_wmma_f32_16x16x32_f16(false, a1, false, b0, (short)0, acc10, false, false);
      acc11 = __builtin_amdgcn_wmma_f32_16x16x32_f16(false, a1, false, b1, (short)0, acc11, false, false);
    }
  }

  // D layout: lane -> N = lane%16; VGPR v -> M = v + 8*(lane/16)
  int n = lane & 15;
  int mBase = (lane >> 4) << 3;
  #pragma unroll
  for (int t = 0; t < 2; ++t) {              // oc half-tile
    const v8f* accs[2] = { t ? &acc10 : &acc00, t ? &acc11 : &acc01 };
    #pragma unroll
    for (int v = 0; v < 8; ++v) {
      int oc = (ocT2 << 5) + (t << 4) + mBase + v;
      if (oc < OC) {
        float bv = bias[oc];
        #pragma unroll
        for (int r = 0; r < 2; ++r) {        // pixel row tile
          float val = (*accs[r])[v] + bv;
          if (RELU) val = fmaxf(val, 0.f);
          size_t o = (size_t)oc * HW + (size_t)(y0 + r) * W + x0 + n;
          if (out16) out16[o] = (half_t)val;
          if (out32) out32[o] = val;
        }
      }
    }
  }
}

// ---------------------------------------------------------------------------
// conv1: IC=1 (K=9) -- direct VALU conv + ReLU -> f16 (0.35 GFLOP, trivial)
// ---------------------------------------------------------------------------
__global__ void conv1_k(const float* __restrict__ img, const float* __restrict__ w,
                        const float* __restrict__ b, half_t* __restrict__ out,
                        int H, int W) {
  int idx = blockIdx.x * blockDim.x + threadIdx.x;
  int HW = H * W;
  if (idx >= 64 * HW) return;
  int oc = idx / HW, p = idx - oc * HW;
  int y = p / W, x = p - y * W;
  float s = b[oc];
  #pragma unroll
  for (int ky = 0; ky < 3; ++ky) {
    int yy = y + ky - 1;
    if (yy < 0 || yy >= H) continue;
    #pragma unroll
    for (int kx = 0; kx < 3; ++kx) {
      int xx = x + kx - 1;
      if (xx < 0 || xx >= W) continue;
      s += img[yy * W + xx] * w[oc * 9 + ky * 3 + kx];
    }
  }
  out[idx] = (half_t)fmaxf(s, 0.f);
}

__global__ void maxpool2_k(const half_t* __restrict__ in, half_t* __restrict__ out,
                           int C, int Ho, int Wo) {
  int idx = blockIdx.x * blockDim.x + threadIdx.x;
  if (idx >= C * Ho * Wo) return;
  int c = idx / (Ho * Wo), p = idx - c * (Ho * Wo);
  int y = p / Wo, x = p - y * Wo;
  int Hi = Ho * 2, Wi = Wo * 2;
  const half_t* s = in + (size_t)c * Hi * Wi + (size_t)(2 * y) * Wi + 2 * x;
  float a0 = (float)s[0], a1 = (float)s[1];
  float a2 = (float)s[Wi], a3 = (float)s[Wi + 1];
  out[idx] = (half_t)fmaxf(fmaxf(a0, a1), fmaxf(a2, a3));
}

// softmax over 65 channels, drop dustbin, pixel-shuffle(8) -> det (480x640)
__global__ void softmax_shuffle_k(const float* __restrict__ logits,
                                  float* __restrict__ det, int h, int w) {
  int p = blockIdx.x * blockDim.x + threadIdx.x;
  int hw = h * w;
  if (p >= hw) return;
  int y = p / w, x = p - y * w;
  float mx = -3.0e38f;
  for (int c = 0; c < 65; ++c) mx = fmaxf(mx, logits[(size_t)c * hw + p]);
  float sum = 0.f;
  for (int c = 0; c < 65; ++c) sum += __expf(logits[(size_t)c * hw + p] - mx);
  float inv = 1.f / sum;
  int W = w * 8;
  for (int c = 0; c < 64; ++c) {
    int i = c >> 3, j = c & 7;
    det[(size_t)(y * 8 + i) * W + (x * 8 + j)] =
        __expf(logits[(size_t)c * hw + p] - mx) * inv;
  }
}

__global__ void hmax_k(const float* __restrict__ in, float* __restrict__ out,
                       int H, int W) {
  int idx = blockIdx.x * blockDim.x + threadIdx.x;
  if (idx >= H * W) return;
  int y = idx / W, x = idx - y * W;
  int lo = max(x - NMSR, 0), hi = min(x + NMSR, W - 1);
  float m = -3.0e38f;
  for (int xx = lo; xx <= hi; ++xx) m = fmaxf(m, in[(size_t)y * W + xx]);
  out[idx] = m;
}

__global__ void vmax_k(const float* __restrict__ in, float* __restrict__ out,
                       int H, int W) {
  int idx = blockIdx.x * blockDim.x + threadIdx.x;
  if (idx >= H * W) return;
  int y = idx / W, x = idx - y * W;
  int lo = max(y - NMSR, 0), hi = min(y + NMSR, H - 1);
  float m = -3.0e38f;
  for (int yy = lo; yy <= hi; ++yy) m = fmaxf(m, in[(size_t)yy * W + x]);
  out[idx] = m;
}

__global__ void cand_k(const float* __restrict__ det, const float* __restrict__ pool,
                       float* __restrict__ cs, int* __restrict__ ci,
                       unsigned int* __restrict__ cnt, int HW) {
  int p = blockIdx.x * blockDim.x + threadIdx.x;
  if (p >= HW) return;
  float s = det[p];
  if (s > THRESH && s == pool[p]) {
    unsigned int slot = atomicAdd(cnt, 1u);
    if (slot < MAXC) { cs[slot] = s; ci[slot] = p; }
  }
}

// Deterministic top-2048 by (score desc, index asc) via pairwise rank.
__global__ void select_k(const float* __restrict__ cs, const int* __restrict__ ci,
                         const unsigned int* __restrict__ cnt,
                         float* __restrict__ kpts, float* __restrict__ scores,
                         int W) {
  int i = blockIdx.x * blockDim.x + threadIdx.x;
  int n = (int)min(*cnt, (unsigned int)MAXC);
  if (i >= n) return;
  float si = cs[i];
  int ii = ci[i];
  int rank = 0;
  for (int j = 0; j < n; ++j) {
    float sj = cs[j];
    int ij = ci[j];
    if (sj > si || (sj == si && ij < ii)) ++rank;
  }
  if (rank < MAXK) {
    scores[rank] = si;
    kpts[rank * 2 + 0] = (float)(ii % W);
    kpts[rank * 2 + 1] = (float)(ii / W);
  }
}

// L2-normalize coarse descriptors (256 ch @ 60x80)
__global__ void norm_desc_k(const float* __restrict__ d, float* __restrict__ dn,
                            int hw) {
  int p = blockIdx.x * blockDim.x + threadIdx.x;
  if (p >= hw) return;
  float ss = 0.f;
  for (int c = 0; c < 256; ++c) { float v = d[(size_t)c * hw + p]; ss += v * v; }
  float inv = 1.f / fmaxf(sqrtf(ss), 1e-12f);
  for (int c = 0; c < 256; ++c) dn[(size_t)c * hw + p] = d[(size_t)c * hw + p] * inv;
}

// Bilinear (align_corners) sample of normalized coarse descs at keypoints only
__global__ void desc_gather_k(const float* __restrict__ dn,
                              const float* __restrict__ kpts,
                              const float* __restrict__ scores,
                              float* __restrict__ out) {
  int k = blockIdx.x;
  int c = threadIdx.x;
  if (scores[k] <= 0.f) return;              // invalid slots stay zeroed
  float x = kpts[k * 2 + 0], y = kpts[k * 2 + 1];
  float xs = x * (79.0f / 639.0f);
  float ys = y * (59.0f / 479.0f);
  int x0 = (int)floorf(xs), y0 = (int)floorf(ys);
  int x1 = min(x0 + 1, 79), y1 = min(y0 + 1, 59);
  float wx = xs - (float)x0, wy = ys - (float)y0;
  const int hw = 60 * 80;
  const float* dc = dn + (size_t)c * hw;
  float d00 = dc[y0 * 80 + x0], d01 = dc[y0 * 80 + x1];
  float d10 = dc[y1 * 80 + x0], d11 = dc[y1 * 80 + x1];
  out[(size_t)k * 256 + c] =
      (d00 * (1.f - wy) + d10 * wy) * (1.f - wx) +
      (d01 * (1.f - wy) + d11 * wy) * wx;
}

// ---------------------------------------------------------------------------
static inline size_t alignUp(size_t x) { return (x + 255) & ~(size_t)255; }

template <int OC, int IC, int H, int W, int K3, int RELU>
static void run_conv(hipStream_t stream, const half_t* in, const float* w,
                     const float* bias, half_t* out16, float* out32,
                     half_t* apack) {
  constexpr int KH = K3 ? 3 : 1;
  constexpr int nKc = IC >> 5, ntaps = KH * KH;
  constexpr int numOcT2 = (OC + 31) >> 5;
  constexpr int numOcTPad = numOcT2 * 2;
  constexpr int packTotal = numOcTPad * ntaps * nKc * 32 * 16;
  pack_weights_k<<<(packTotal + 255) / 256, 256, 0, stream>>>(w, apack, OC, IC,
                                                              KH, KH, packTotal);
  constexpr int totalTiles = numOcT2 * (H >> 1) * (W >> 4);
  constexpr int blocks = (totalTiles + 7) / 8;   // 8 waves per 256-thread block
  conv_wmma_k<OC, IC, H, W, K3, RELU><<<blocks, 256, 0, stream>>>(
      in, apack, bias, out16, out32);
}

extern "C" void kernel_launch(void* const* d_in, const int* in_sizes, int n_in,
                              void* d_out, int out_size, void* d_ws, size_t ws_size,
                              hipStream_t stream) {
  const float* image = (const float*)d_in[0];
  const float* we1 = (const float*)d_in[1];  const float* be1 = (const float*)d_in[2];
  const float* we2 = (const float*)d_in[3];  const float* be2 = (const float*)d_in[4];
  const float* we3 = (const float*)d_in[5];  const float* be3 = (const float*)d_in[6];
  const float* we4 = (const float*)d_in[7];  const float* be4 = (const float*)d_in[8];
  const float* we5 = (const float*)d_in[9];  const float* be5 = (const float*)d_in[10];
  const float* we6 = (const float*)d_in[11]; const float* be6 = (const float*)d_in[12];
  const float* we7 = (const float*)d_in[13]; const float* be7 = (const float*)d_in[14];
  const float* we8 = (const float*)d_in[15]; const float* be8 = (const float*)d_in[16];
  const float* wd1 = (const float*)d_in[17]; const float* bd1 = (const float*)d_in[18];
  const float* wd2 = (const float*)d_in[19]; const float* bd2 = (const float*)d_in[20];
  const float* ws1 = (const float*)d_in[21]; const float* bs1 = (const float*)d_in[22];
  const float* ws2 = (const float*)d_in[23]; const float* bs2 = (const float*)d_in[24];

  const int H = 480, W = 640, h = 60, w = 80;
  const size_t HW = (size_t)H * W, hw = (size_t)h * w;

  // workspace layout
  char* ws = (char*)d_ws;
  size_t off = 0;
  half_t* B1 = (half_t*)(ws + off); off += alignUp(64 * HW * sizeof(half_t));   // 39.3 MB
  half_t* B2 = (half_t*)(ws + off); off += alignUp(64 * HW * sizeof(half_t));   // 39.3 MB
  half_t* AP = (half_t*)(ws + off); off += alignUp(2u << 20);                   // 2 MB
  float* DET65 = (float*)(ws + off); off += alignUp(65 * hw * sizeof(float));
  float* S2F   = (float*)(ws + off); off += alignUp(256 * hw * sizeof(float));
  float* DESCN = (float*)(ws + off); off += alignUp(256 * hw * sizeof(float));
  float* DET   = (float*)(ws + off); off += alignUp(HW * sizeof(float));
  float* TMP   = (float*)(ws + off); off += alignUp(HW * sizeof(float));
  float* POOL  = (float*)(ws + off); off += alignUp(HW * sizeof(float));
  float* CS    = (float*)(ws + off); off += alignUp(MAXC * sizeof(float));
  int*   CI    = (int*)(ws + off);   off += alignUp(MAXC * sizeof(int));
  unsigned int* CNT = (unsigned int*)(ws + off); off += alignUp(sizeof(unsigned int));
  (void)ws_size; (void)in_sizes; (void)n_in;

  float* outKpts   = (float*)d_out;              // (2048,2)
  float* outDesc   = outKpts + MAXK * 2;         // (2048,256)
  float* outScores = outDesc + MAXK * 256;       // (2048,)
  hipMemsetAsync(d_out, 0, (size_t)out_size * sizeof(float), stream);
  hipMemsetAsync(CNT, 0, sizeof(unsigned int), stream);

  // --- encoder ---
  conv1_k<<<(int)((64 * HW + 255) / 256), 256, 0, stream>>>(image, we1, be1, B1, H, W);
  run_conv<64, 64, 480, 640, 1, 1>(stream, B1, we2, be2, B2, nullptr, AP);
  maxpool2_k<<<(int)((64 * 240 * 320 + 255) / 256), 256, 0, stream>>>(B2, B1, 64, 240, 320);
  run_conv<64, 64, 240, 320, 1, 1>(stream, B1, we3, be3, B2, nullptr, AP);
  run_conv<64, 64, 240, 320, 1, 1>(stream, B2, we4, be4, B1, nullptr, AP);
  maxpool2_k<<<(int)((64 * 120 * 160 + 255) / 256), 256, 0, stream>>>(B1, B2, 64, 120, 160);
  run_conv<128, 64, 120, 160, 1, 1>(stream, B2, we5, be5, B1, nullptr, AP);
  run_conv<128, 128, 120, 160, 1, 1>(stream, B1, we6, be6, B2, nullptr, AP);
  maxpool2_k<<<(int)((128 * 60 * 80 + 255) / 256), 256, 0, stream>>>(B2, B1, 128, 60, 80);
  run_conv<128, 128, 60, 80, 1, 1>(stream, B1, we7, be7, B2, nullptr, AP);
  run_conv<128, 128, 60, 80, 1, 1>(stream, B2, we8, be8, B1, nullptr, AP);
  // B1 now holds features (128 x 60 x 80, f16)

  // --- detector head ---
  run_conv<256, 128, 60, 80, 1, 1>(stream, B1, wd1, bd1, B2, nullptr, AP);
  run_conv<65, 256, 60, 80, 0, 0>(stream, B2, wd2, bd2, nullptr, DET65, AP);
  softmax_shuffle_k<<<(int)((hw + 255) / 256), 256, 0, stream>>>(DET65, DET, h, w);

  // --- descriptor head ---
  run_conv<256, 128, 60, 80, 1, 1>(stream, B1, ws1, bs1, B2, nullptr, AP);
  run_conv<256, 256, 60, 80, 0, 0>(stream, B2, ws2, bs2, nullptr, S2F, AP);
  norm_desc_k<<<(int)((hw + 255) / 256), 256, 0, stream>>>(S2F, DESCN, (int)hw);

  // --- NMS + top-k ---
  hmax_k<<<(int)((HW + 255) / 256), 256, 0, stream>>>(DET, TMP, H, W);
  vmax_k<<<(int)((HW + 255) / 256), 256, 0, stream>>>(TMP, POOL, H, W);
  cand_k<<<(int)((HW + 255) / 256), 256, 0, stream>>>(DET, POOL, CS, CI, CNT, (int)HW);
  select_k<<<(MAXC + 255) / 256, 256, 0, stream>>>(CS, CI, CNT, outKpts, outScores, W);

  // --- descriptors at keypoints only (skips 315 MB full-res upsample) ---
  desc_gather_k<<<MAXK, 256, 0, stream>>>(DESCN, outKpts, outScores, outDesc);
}
